// LinearAttention_27496380629324
// MI455X (gfx1250) — matrix operand order repaired and checked by reading, not compiled, not run on hardware
//
#include <hip/hip_runtime.h>
#include <hip/hip_bf16.h>

// ---------------------------------------------------------------------------
// CDNA5 (gfx1250) wave32 WMMA kernels for chunked linear attention.
// D = A(16x32 f16) x B(32x16 f16) + C(16x16 f32) via v_wmma_f32_16x16x32_f16.
// Ping-pong register double buffering (peeled, branch-free) so loads overlap
// WMMAs without rotate-copy overhead.
// ---------------------------------------------------------------------------

typedef __attribute__((ext_vector_type(16))) _Float16 v16h;
typedef __attribute__((ext_vector_type(8)))  _Float16 v8h;
typedef __attribute__((ext_vector_type(8)))  float    v8f;

#define DEV static __device__ __forceinline__

union Frag16 { v16h v; v8h h[2]; };

DEV v8f zero8() {
  v8f z = {0.f, 0.f, 0.f, 0.f, 0.f, 0.f, 0.f, 0.f};
  return z;
}

// A-matrix fragment (16x32, 16-bit). Lanes 0-15 row M=lane hold
// K={0..7,16..23}; lanes 16-31 row M=lane-16 hold K={8..15,24..31}.
DEV v16h load_a_frag(const _Float16* tile, int ld) {
  const int lane = threadIdx.x & 31;
  const _Float16* p = tile + (size_t)(lane & 15) * ld + ((lane >> 4) << 3);
  Frag16 f;
  f.h[0] = *reinterpret_cast<const v8h*>(p);
  f.h[1] = *reinterpret_cast<const v8h*>(p + 16);
  return f.v;
}

// B-matrix fragment (32x16, 16-bit). Lane n<16 holds column n, K={0..7,16..23};
// lane n+16 holds column n, K={8..15,24..31}. `tile` points at the row of
// the source matrix whose rows are B's columns (B = W^T with W row-major).
DEV v16h load_b_frag(const _Float16* tile, int ld) {
  const int lane = threadIdx.x & 31;
  const _Float16* p = tile + (size_t)(lane & 15) * ld + ((lane >> 4) << 3);
  Frag16 f;
  f.h[0] = *reinterpret_cast<const v8h*>(p);
  f.h[1] = *reinterpret_cast<const v8h*>(p + 16);
  return f.v;
}

DEV v8f wmma16(v16h a, v16h b, v8f c) {
  return __builtin_amdgcn_wmma_f32_16x16x32_f16(
      /*neg_a=*/false, a, /*neg_b=*/false, b,
      /*c_mod=*/(short)0, c, /*reuse_a=*/false, /*reuse_b=*/false);
}

// ---------------------------------------------------------------------------
// f32 -> f16 conversion (all element counts are multiples of 1024)
// ---------------------------------------------------------------------------
__global__ __launch_bounds__(256) void cvt_kernel(
    const float* __restrict__ s, _Float16* __restrict__ d, long long n) {
  long long i = ((long long)blockIdx.x * 256 + threadIdx.x) * 4;
  if (i < n) {
    float4 f = *reinterpret_cast<const float4*>(s + i);
    d[i + 0] = (_Float16)f.x;
    d[i + 1] = (_Float16)f.y;
    d[i + 2] = (_Float16)f.z;
    d[i + 3] = (_Float16)f.w;
  }
}

// ---------------------------------------------------------------------------
// GEMM: out[M,N] = X[M,K] @ W[N,K]^T, f16 in, f32 accumulate.
// Block: 256 threads (8 waves), block tile 256(M) x 64(N); each wave owns a
// 32x64 sub-tile (2 m-frags x 4 n-frags -> 8 WMMAs per 6 fragment loads).
// Peeled ping-pong k-loop: buf0 = even k-steps, buf1 = odd k-steps.
// mode 0: store f16 row-major (M,N)
// mode 1: store f16 transposed per head: VT[b,h,dv,t]   (v projection)
// mode 2: store f32 row-major (final output projection)
// ---------------------------------------------------------------------------
__global__ __launch_bounds__(256) void gemm_nt_kernel(
    const _Float16* __restrict__ X, const _Float16* __restrict__ W,
    void* __restrict__ out, int M, int N, int K, int mode, int T, int H) {
  const int w = threadIdx.x >> 5;
  const int lane = threadIdx.x & 31;
  const int m0 = blockIdx.x * 256 + w * 32;
  const int n0 = blockIdx.y * 64;

  v8f acc[2][4];
#pragma unroll
  for (int mi = 0; mi < 2; ++mi)
#pragma unroll
    for (int nj = 0; nj < 4; ++nj) acc[mi][nj] = zero8();

  const _Float16* Ap = X + (size_t)m0 * K;
  const _Float16* Bp = W + (size_t)n0 * K;

  v16h a0[2], b0[4], a1[2], b1[4];
  auto loadA = [&](v16h* a, int kk) {
    a[0] = load_a_frag(Ap + kk, K);
    a[1] = load_a_frag(Ap + (size_t)16 * K + kk, K);
  };
  auto loadB = [&](v16h* bf, int kk) {
#pragma unroll
    for (int nj = 0; nj < 4; ++nj)
      bf[nj] = load_b_frag(Bp + (size_t)(nj * 16) * K + kk, K);
  };
  auto mulacc = [&](v16h* a, v16h* bf) {
#pragma unroll
    for (int nj = 0; nj < 4; ++nj) {
      acc[0][nj] = wmma16(a[0], bf[nj], acc[0][nj]);
      acc[1][nj] = wmma16(a[1], bf[nj], acc[1][nj]);
    }
  };

  loadA(a0, 0);
  loadB(b0, 0);
  loadA(a1, 32);
  loadB(b1, 32);
  for (int kk = 0; kk + 64 < K; kk += 64) {
    mulacc(a0, b0);
    loadA(a0, kk + 64);
    loadB(b0, kk + 64);
    mulacc(a1, b1);
    loadA(a1, kk + 96);
    loadB(b1, kk + 96);
  }
  mulacc(a0, b0);
  mulacc(a1, b1);

  const int col = lane & 15, hi = lane >> 4;
#pragma unroll
  for (int mi = 0; mi < 2; ++mi) {
#pragma unroll
    for (int nj = 0; nj < 4; ++nj) {
#pragma unroll
      for (int r = 0; r < 8; ++r) {
        int gm = m0 + mi * 16 + r + 8 * hi;
        int gn = n0 + nj * 16 + col;
        float val = acc[mi][nj][r];
        if (mode == 0) {
          ((_Float16*)out)[(size_t)gm * N + gn] = (_Float16)val;
        } else if (mode == 1) {
          int bb = gm / T, t = gm - bb * T;
          int hh = gn >> 7, dv = gn & 127;
          ((_Float16*)out)[(((size_t)bb * H + hh) * 128 + dv) * T + t] =
              (_Float16)val;
        } else {
          ((float*)out)[(size_t)gm * N + gn] = val;
        }
      }
    }
  }
}

// ---------------------------------------------------------------------------
// Hadamard feature map: out[m,n] = (X@W1^T + b1)[m,n] * (X@W2^T + b2)[m,n] * s
// X viewed as (M2 = B*T*H, 128). Optionally also store transposed per head:
// outT[b,h,n,t]  (used for phi_k -> kc^T GEMM).
// Block tile 64(M) x 128(N), 8 waves: wave w -> m-tile (w&3), n-tiles
// [(w>>2)*4, +4). A-fragment ping-ponged; weight B-frags stay L2-hot.
// ---------------------------------------------------------------------------
__global__ __launch_bounds__(256) void feature_map_kernel(
    const _Float16* __restrict__ Xin,
    const _Float16* __restrict__ W1, const _Float16* __restrict__ W2,
    const float* __restrict__ b1, const float* __restrict__ b2, float scale,
    _Float16* __restrict__ outN, _Float16* __restrict__ outT, int T, int H) {
  const int w = threadIdx.x >> 5, lane = threadIdx.x & 31;
  const int mi = w & 3, njb = (w >> 2) * 4;
  const int m0 = blockIdx.x * 64 + mi * 16;

  v8f a1[4], a2[4];
#pragma unroll
  for (int j = 0; j < 4; ++j) { a1[j] = zero8(); a2[j] = zero8(); }

  const _Float16* Xp = Xin + (size_t)m0 * 128;
  auto step = [&](v16h av, int kk) {
#pragma unroll
    for (int j = 0; j < 4; ++j) {
      int nrow = (njb + j) * 16;
      v16h bf1 = load_b_frag(W1 + (size_t)nrow * 128 + kk, 128);
      a1[j] = wmma16(av, bf1, a1[j]);
      v16h bf2 = load_b_frag(W2 + (size_t)nrow * 128 + kk, 128);
      a2[j] = wmma16(av, bf2, a2[j]);
    }
  };

  v16h aA = load_a_frag(Xp, 128);
  v16h aB = load_a_frag(Xp + 32, 128);
  step(aA, 0);
  aA = load_a_frag(Xp + 64, 128);
  step(aB, 32);
  aB = load_a_frag(Xp + 96, 128);
  step(aA, 64);
  step(aB, 96);

  const int col = lane & 15, hi = lane >> 4;
#pragma unroll
  for (int j = 0; j < 4; ++j) {
    int gn = (njb + j) * 16 + col;
    float bb1 = b1[gn], bb2 = b2[gn];
#pragma unroll
    for (int r = 0; r < 8; ++r) {
      int gm = m0 + r + 8 * hi;
      float val = (a1[j][r] + bb1) * (a2[j][r] + bb2) * scale;
      outN[(size_t)gm * 128 + gn] = (_Float16)val;
      if (outT) {
        int bb = gm / (T * H);
        int rem = gm - bb * (T * H);
        int t = rem / H;
        int hh = rem - t * H;
        outT[(((size_t)bb * H + hh) * 128 + gn) * T + t] = (_Float16)val;
      }
    }
  }
}

// ---------------------------------------------------------------------------
// Per-chunk state: KV[bh,n,k,v] = sum_c kc[c,k] * vc[c,v]
// A = KT rows (k-dim major, contiguous in t), B columns = VT rows.
// Block per (bh,n); wave w owns k-rows [w*16, +16); B-frags ping-ponged
// across the 8 v-tiles.
// ---------------------------------------------------------------------------
__global__ __launch_bounds__(256) void chunk_kv_kernel(
    const _Float16* __restrict__ KT, const _Float16* __restrict__ VT,
    float* __restrict__ KV, int T, int N) {
  const int w = threadIdx.x >> 5, lane = threadIdx.x & 31;
  const int n = blockIdx.x % N;
  const int bh = blockIdx.x / N;
  const int t0 = n * 64;
  const size_t headoff = (size_t)bh * 128 * T;
  const int col = lane & 15, hi = lane >> 4;

  v16h a0 = load_a_frag(KT + headoff + (size_t)(w * 16) * T + t0, T);
  v16h a1 = load_a_frag(KT + headoff + (size_t)(w * 16) * T + t0 + 32, T);

  const size_t base = ((size_t)bh * N + n) * 128 * 128;
  auto loadB2 = [&](int nj, v16h& x, v16h& y) {
    x = load_b_frag(VT + headoff + (size_t)(nj * 16) * T + t0, T);
    y = load_b_frag(VT + headoff + (size_t)(nj * 16) * T + t0 + 32, T);
  };
  auto compute = [&](int nj, v16h x, v16h y) {
    v8f acc = zero8();
    acc = wmma16(a0, x, acc);
    acc = wmma16(a1, y, acc);
#pragma unroll
    for (int r = 0; r < 8; ++r) {
      int krow = w * 16 + r + 8 * hi;
      int vcol = nj * 16 + col;
      KV[base + (size_t)krow * 128 + vcol] = acc[r];
    }
  };

  v16h p0a, p0b, p1a, p1b;
  loadB2(0, p0a, p0b);
  loadB2(1, p1a, p1b);
#pragma unroll
  for (int nj = 0; nj < 6; nj += 2) {
    compute(nj, p0a, p0b);
    loadB2(nj + 2, p0a, p0b);
    compute(nj + 1, p1a, p1b);
    loadB2(nj + 3, p1a, p1b);
  }
  compute(6, p0a, p0b);
  compute(7, p1a, p1b);
}

// ---------------------------------------------------------------------------
// Exclusive cumsum over chunks; writes the state TRANSPOSED as f16:
// ST[bh,n,v,k] = sum_{n'<n} KV[bh,n',k,v].
// ---------------------------------------------------------------------------
__global__ __launch_bounds__(256) void cumsum_kernel(
    const float* __restrict__ KV, _Float16* __restrict__ ST, int N) {
  const int bh = blockIdx.x >> 6;
  const int part = blockIdx.x & 63;
  const int p = part * 256 + threadIdx.x;
  const int v = p >> 7, k = p & 127;
  float acc = 0.f;
  const size_t base = (size_t)bh * N * 16384;
  for (int n = 0; n < N; ++n) {
    size_t slab = base + (size_t)n * 16384;
    ST[slab + (size_t)v * 128 + k] = (_Float16)acc;
    acc += KV[slab + (size_t)k * 128 + v];
  }
}

// ---------------------------------------------------------------------------
// Attention per (b,h,chunk):
//  Phase A: Afull(64x64) = tril(qc @ kc^T)  -> masked, f16, staged in LDS
//  Phase B: O(64x128) = Afull @ vc + qc @ S_excl
// qc carries the 1/sqrt(Dk) scale (folded into phi_q). Phase-B A-fragments
// are hoisted (reused by all 4 v-tiles); B-fragment sets ping-ponged.
// ---------------------------------------------------------------------------
__global__ __launch_bounds__(256) void attn_kernel(
    const _Float16* __restrict__ PHIQ,  // (B,T,H,128)
    const _Float16* __restrict__ PHIK,  // (B,T,H,128)
    const _Float16* __restrict__ VT,    // (B*H,128,T)
    const _Float16* __restrict__ ST,    // (B*H,N,128[v],128[k]) f16
    float* __restrict__ O,              // (B,T,H,128)
    int T, int H, int N) {
  __shared__ _Float16 smA[64 * 80];  // 64x64 tile, padded stride (16B aligned)
  const int w = threadIdx.x >> 5, lane = threadIdx.x & 31;
  const int n = blockIdx.x % N;
  const int h = (blockIdx.x / N) % H;
  const int b = blockIdx.x / (N * H);
  const int t0 = n * 64;
  const int col = lane & 15, hi = lane >> 4;
  const int ldqk = H * 128;
  const size_t qkRow0 = ((size_t)(b * T + t0) * H + h) * 128;

  // ---- Phase A: scores, causal mask, stage to LDS (ping-pong k-loop) ----
#pragma unroll
  for (int tix = 0; tix < 2; ++tix) {
    int tid2 = w * 2 + tix;
    int ti = tid2 >> 2, tj = tid2 & 3;
    const _Float16* qb = PHIQ + qkRow0 + (size_t)(ti * 16) * ldqk;
    const _Float16* kb = PHIK + qkRow0 + (size_t)(tj * 16) * ldqk;
    v16h aA = load_a_frag(qb, ldqk);
    v16h bA = load_b_frag(kb, ldqk);
    v16h aB = load_a_frag(qb + 32, ldqk);
    v16h bB = load_b_frag(kb + 32, ldqk);
    v8f acc = zero8();
    acc = wmma16(aA, bA, acc);
    aA = load_a_frag(qb + 64, ldqk);
    bA = load_b_frag(kb + 64, ldqk);
    acc = wmma16(aB, bB, acc);
    aB = load_a_frag(qb + 96, ldqk);
    bB = load_b_frag(kb + 96, ldqk);
    acc = wmma16(aA, bA, acc);
    acc = wmma16(aB, bB, acc);
#pragma unroll
    for (int r = 0; r < 8; ++r) {
      int c = ti * 16 + r + 8 * hi;
      int s = tj * 16 + col;
      float val = (s <= c) ? acc[r] : 0.f;
      smA[c * 80 + s] = (_Float16)val;
    }
  }
  __syncthreads();

  // ---- Phase B: O = A @ vc + qc @ S_excl ----
  const int mi = w >> 1;
  const int njb = (w & 1) * 4;
  const size_t headoff = ((size_t)b * H + h) * 128 * T;
  const size_t stoff = (((size_t)b * H + h) * N + n) * 128 * 128;

  // Hoisted A fragments: reused across all 4 v-tiles of this wave.
  v16h aI0 = load_a_frag(smA + mi * 16 * 80, 80);
  v16h aI1 = load_a_frag(smA + mi * 16 * 80 + 32, 80);
  v16h aS[4];
#pragma unroll
  for (int q = 0; q < 4; ++q)
    aS[q] = load_a_frag(PHIQ + qkRow0 + (size_t)(mi * 16) * ldqk + q * 32, ldqk);

  auto loadBs = [&](int nj, v16h* bv) {
    bv[0] = load_b_frag(VT + headoff + (size_t)(nj * 16) * T + t0, T);
    bv[1] = load_b_frag(VT + headoff + (size_t)(nj * 16) * T + t0 + 32, T);
#pragma unroll
    for (int q = 0; q < 4; ++q)
      bv[2 + q] =
          load_b_frag(ST + stoff + (size_t)(nj * 16) * 128 + q * 32, 128);
  };
  auto computeJ = [&](int nj, v16h* bv) {
    v8f acc = zero8();
    acc = wmma16(aI0, bv[0], acc);
    acc = wmma16(aI1, bv[1], acc);
#pragma unroll
    for (int q = 0; q < 4; ++q) acc = wmma16(aS[q], bv[2 + q], acc);
#pragma unroll
    for (int r = 0; r < 8; ++r) {
      int t = t0 + mi * 16 + r + 8 * hi;
      int vcol = nj * 16 + col;
      O[((size_t)(b * T + t) * H + h) * 128 + vcol] = acc[r];
    }
  };

  v16h s0[6], s1[6];
  loadBs(njb + 0, s0);
  loadBs(njb + 1, s1);
  computeJ(njb + 0, s0);
  loadBs(njb + 2, s0);
  computeJ(njb + 1, s1);
  loadBs(njb + 3, s1);
  computeJ(njb + 2, s0);
  computeJ(njb + 3, s1);
}

// ---------------------------------------------------------------------------
// Per-head RMSNorm over Dh=128, f32 in -> f16 out. One wave32 per row.
// ---------------------------------------------------------------------------
__global__ __launch_bounds__(256) void rmsnorm_kernel(
    const float* __restrict__ X, const float* __restrict__ wgt,
    _Float16* __restrict__ Y) {
  const int wv = threadIdx.x >> 5, lane = threadIdx.x & 31;
  const int row = blockIdx.x * 8 + wv;
  const float* x = X + (size_t)row * 128;
  float v[4], ss = 0.f;
#pragma unroll
  for (int i = 0; i < 4; ++i) {
    v[i] = x[lane + 32 * i];
    ss += v[i] * v[i];
  }
#pragma unroll
  for (int off = 16; off > 0; off >>= 1) ss += __shfl_xor(ss, off, 32);
  float inv = rsqrtf(ss * (1.f / 128.f) + 1e-5f);
#pragma unroll
  for (int i = 0; i < 4; ++i) {
    int j = lane + 32 * i;
    Y[(size_t)row * 128 + j] = (_Float16)(v[i] * inv * wgt[j]);
  }
}

// ---------------------------------------------------------------------------
extern "C" void kernel_launch(void* const* d_in, const int* in_sizes, int n_in,
                              void* d_out, int out_size, void* d_ws,
                              size_t ws_size, hipStream_t stream) {
  (void)in_sizes; (void)n_in; (void)out_size; (void)ws_size;
  constexpr int B = 2, T = 8192, D = 1024, H = 8, Dh = 128;
  constexpr int N = 128;                      // T / CHUNK
  constexpr long long MT = (long long)B * T;  // 16384 rows
  constexpr long long M2 = MT * H;            // 131072 per-head rows

  const float* hs   = (const float*)d_in[0];
  const float* qw   = (const float*)d_in[2];
  const float* kw   = (const float*)d_in[3];
  const float* vw   = (const float*)d_in[4];
  const float* ow   = (const float*)d_in[5];
  const float* fq1  = (const float*)d_in[6];
  const float* fqb1 = (const float*)d_in[7];
  const float* fq2  = (const float*)d_in[8];
  const float* fqb2 = (const float*)d_in[9];
  const float* fk1  = (const float*)d_in[10];
  const float* fkb1 = (const float*)d_in[11];
  const float* fk2  = (const float*)d_in[12];
  const float* fkb2 = (const float*)d_in[13];
  const float* rmsw = (const float*)d_in[14];

  // Workspace layout (slots reused across pipeline stages):
  char* ws = (char*)d_ws;
  const size_t S16 = (size_t)MT * D * sizeof(_Float16);  // 33.5 MB
  _Float16* slotA = (_Float16*)(ws);            // HS16  -> PHIQ
  _Float16* slotB = (_Float16*)(ws + 1 * S16);  // Q16   -> PHIK
  _Float16* slotC = (_Float16*)(ws + 2 * S16);  // K16   -> O16 (post-norm)
  _Float16* slotD = (_Float16*)(ws + 3 * S16);  // VT (B*H,128,T)
  _Float16* slotE = (_Float16*)(ws + 4 * S16);  // PHIKT (B*H,128,T)
  _Float16* w16   = (_Float16*)(ws + 5 * S16);  // f16 weights
  _Float16* qw16  = w16;
  _Float16* kw16  = w16 + (size_t)D * D;
  _Float16* vw16  = w16 + 2 * (size_t)D * D;
  _Float16* ow16  = w16 + 3 * (size_t)D * D;
  _Float16* fq116 = w16 + 4 * (size_t)D * D;
  _Float16* fq216 = fq116 + Dh * Dh;
  _Float16* fk116 = fq216 + Dh * Dh;
  _Float16* fk216 = fk116 + Dh * Dh;
  size_t woff = 5 * S16 + (((4 * (size_t)D * D + 4 * (size_t)Dh * Dh) *
                                sizeof(_Float16) + 255) & ~(size_t)255);
  float*    KV  = (float*)(ws + woff);  // 134 MB
  const size_t SKV = (size_t)B * H * N * 128 * 128 * sizeof(float);
  _Float16* STt = (_Float16*)(ws + woff + SKV);  // 67 MB
  const size_t SST = (size_t)B * H * N * 128 * 128 * sizeof(_Float16);
  float*    O32 = (float*)(ws + woff + SKV + SST);  // 67 MB

  // 1) f32 -> f16 conversions
  cvt_kernel<<<(int)(MT * D / 1024), 256, 0, stream>>>(hs, slotA, MT * D);
  cvt_kernel<<<D * D / 1024, 256, 0, stream>>>(qw, qw16, (long long)D * D);
  cvt_kernel<<<D * D / 1024, 256, 0, stream>>>(kw, kw16, (long long)D * D);
  cvt_kernel<<<D * D / 1024, 256, 0, stream>>>(vw, vw16, (long long)D * D);
  cvt_kernel<<<D * D / 1024, 256, 0, stream>>>(ow, ow16, (long long)D * D);
  cvt_kernel<<<Dh * Dh / 1024, 256, 0, stream>>>(fq1, fq116, Dh * Dh);
  cvt_kernel<<<Dh * Dh / 1024, 256, 0, stream>>>(fq2, fq216, Dh * Dh);
  cvt_kernel<<<Dh * Dh / 1024, 256, 0, stream>>>(fk1, fk116, Dh * Dh);
  cvt_kernel<<<Dh * Dh / 1024, 256, 0, stream>>>(fk2, fk216, Dh * Dh);

  // 2) Q/K/V projections (block tile 256x64)
  dim3 gProj((unsigned)(MT / 256), (unsigned)(D / 64));
  gemm_nt_kernel<<<gProj, 256, 0, stream>>>(slotA, qw16, slotB, (int)MT, D, D,
                                            0, T, H);
  gemm_nt_kernel<<<gProj, 256, 0, stream>>>(slotA, kw16, slotC, (int)MT, D, D,
                                            0, T, H);
  gemm_nt_kernel<<<gProj, 256, 0, stream>>>(slotA, vw16, slotD, (int)MT, D, D,
                                            1, T, H);

  // 3) feature maps (q gets the 1/sqrt(Dk) scale folded in)
  const float scale = 0.08838834764831845f;  // 128^-0.5
  feature_map_kernel<<<(int)(M2 / 64), 256, 0, stream>>>(
      slotB, fq116, fq216, fqb1, fqb2, scale, slotA, nullptr, T, H);
  feature_map_kernel<<<(int)(M2 / 64), 256, 0, stream>>>(
      slotC, fk116, fk216, fkb1, fkb2, 1.0f, slotB, slotE, T, H);

  // 4) per-chunk kv outer products, then exclusive prefix state (transposed)
  chunk_kv_kernel<<<B * H * N, 256, 0, stream>>>(slotE, slotD, KV, T, N);
  cumsum_kernel<<<B * H * 64, 256, 0, stream>>>(KV, STt, N);

  // 5) intra + inter chunk attention
  attn_kernel<<<B * H * N, 256, 0, stream>>>(slotA, slotB, slotD, STt, O32, T,
                                             H, N);

  // 6) per-head RMSNorm, then output projection (f32 out)
  rmsnorm_kernel<<<(int)(M2 / 8), 256, 0, stream>>>(O32, rmsw, slotC);
  gemm_nt_kernel<<<gProj, 256, 0, stream>>>(slotC, ow16, d_out, (int)MT, D, D,
                                            2, T, H);
}
